// CostAggregation_11_58780922413129
// MI455X (gfx1250) — matrix operand order repaired and checked by reading, not compiled, not run on hardware
//
#include <hip/hip_runtime.h>
#include <hip/hip_bf16.h>
#include <stdint.h>

// ============================================================================
// CDNA5 / gfx1250 implementation of the GA-Net style 3D cost aggregation net.
//  - 3x3x3 convs: implicit-GEMM via v_wmma_f32_16x16x32_bf16 (bf16 in, f32 acc)
//    16(M=cout) x 32(N=w outputs) tile per wave, K = Cin*27 chunked by 32
//  - activations kept in channel-interleaved bf16 [cblock][d][h][w][32ch] so
//    input tiles stage to LDS as contiguous global_load_async_to_lds_b128 and
//    B fragments are built with 2x ds_load_b128 (channels = K innermost)
//  - per-(kd,kh) weight slabs staged to LDS by the Tensor Data Mover
//    (TENSOR_LOAD_TO_LDS + s_wait_tensorcnt), with global fallback
//  - SGA scans: wave32 recurrence, lane == D, shfl up/down/xor for D coupling
//  - deconvs: direct VALU kernels (tiny FLOP share)
// ============================================================================

#define USE_ASYNC_STAGE 1

#if defined(__HIP_DEVICE_COMPILE__) && \
    __has_builtin(__builtin_amdgcn_tensor_load_to_lds) && \
    __has_builtin(__builtin_amdgcn_s_wait_tensorcnt)
#define HAVE_TDM 1
#else
#define HAVE_TDM 0
#endif

typedef __attribute__((ext_vector_type(16))) __bf16 v16bf;
typedef __attribute__((ext_vector_type(8)))  float  v8f;
typedef unsigned int u32x4 __attribute__((ext_vector_type(4)));
typedef int          i32x8 __attribute__((ext_vector_type(8)));
typedef int          i32x4 __attribute__((ext_vector_type(4)));

__device__ __forceinline__ unsigned short f2bf(float f) {
  uint32_t u = __builtin_bit_cast(uint32_t, f);
  uint32_t r = (u + 0x7fffu + ((u >> 16) & 1u)) >> 16;   // round-to-nearest-even
  return (unsigned short)r;
}

// ---------------------------------------------------------------------------
// Pre-pack OIDHW fp32 weights into bf16 WMMA A-fragments.
// Fragment index: ((tap*nCh + chunk)*MT + mtile)*32 + lane, 8 dwords each.
// A layout (16-bit 16x32, ISA 7.12.2): lane<16 -> M=lane, K pairs
//   v0..3 = K(0,1)(2,3)(4,5)(6,7), v4..7 = K(16..23);
// lane>=16 -> M=lane-16, K offset +8 (v0..3 -> 8..15, v4..7 -> 24..31).
// ---------------------------------------------------------------------------
__global__ void prepack_w_kernel(const float* __restrict__ wsrc,
                                 uint32_t* __restrict__ wpk,
                                 int Cin, int Cout) {
  int nCh = (Cin + 31) >> 5;
  int MT  = Cout >> 4;
  int total = 27 * nCh * MT * 32;
  int t = blockIdx.x * blockDim.x + threadIdx.x;
  if (t >= total) return;
  int lane = t & 31;
  int rest = t >> 5;
  int mt = rest % MT; rest /= MT;
  int ch = rest % nCh; rest /= nCh;
  int tap = rest;                      // (kd*3+kh)*3+kw
  int m = lane & 15;
  int khalf = (lane >> 4) ? 8 : 0;
  int cout = mt * 16 + m;
  uint32_t out[8];
#pragma unroll
  for (int j = 0; j < 8; ++j) {
    int koff = ((j < 4) ? (2 * j) : (16 + 2 * (j - 4))) + khalf;
    int c0 = ch * 32 + koff;
    unsigned short lo = 0, hi = 0;
    if (c0 < Cin)     lo = f2bf(wsrc[((size_t)cout * Cin + c0) * 27 + tap]);
    if (c0 + 1 < Cin) hi = f2bf(wsrc[((size_t)cout * Cin + c0 + 1) * 27 + tap]);
    out[j] = (uint32_t)lo | ((uint32_t)hi << 16);
  }
  uint32_t* dst = wpk + (size_t)t * 8;
#pragma unroll
  for (int j = 0; j < 8; ++j) dst[j] = out[j];
}

// ---------------------------------------------------------------------------
// fp32 NCDHW -> channel-interleaved bf16 [cblock][d][h][w][32c].
// 32(c) x 32(w) tile transpose through LDS; zero-fills c >= Cin.
// ---------------------------------------------------------------------------
#define CVT_RS 36

__global__ void cvt_interleave_kernel(const float* __restrict__ in,
                                      unsigned short* __restrict__ outp,
                                      int Cin, int Di, int Hi, int Wi) {
  __shared__ unsigned short tile[32 * CVT_RS];
  int nW = Wi >> 5;
  int b = blockIdx.x;
  int wt = b % nW; int rest = b / nW;
  int h = rest % Hi; rest /= Hi;
  int d = rest % Di; int cb = rest / Di;
  int w0 = wt << 5;
  int t = threadIdx.x;
  {
    int c  = t >> 3;
    int wq = (t & 7) << 2;
    int cg = cb * 32 + c;
    float4 f = {0.f, 0.f, 0.f, 0.f};
    if (cg < Cin)
      f = *(const float4*)(in + (((size_t)cg * Di + d) * Hi + h) * (size_t)Wi + w0 + wq);
    unsigned short* tp = &tile[c * CVT_RS + wq];
    tp[0] = f2bf(f.x); tp[1] = f2bf(f.y); tp[2] = f2bf(f.z); tp[3] = f2bf(f.w);
  }
  __syncthreads();
  {
    int w  = t >> 3;
    int cq = (t & 7) << 2;
    uint32_t d0 = (uint32_t)tile[(cq + 0) * CVT_RS + w] |
                  ((uint32_t)tile[(cq + 1) * CVT_RS + w] << 16);
    uint32_t d1 = (uint32_t)tile[(cq + 2) * CVT_RS + w] |
                  ((uint32_t)tile[(cq + 3) * CVT_RS + w] << 16);
    size_t obase = ((((size_t)cb * Di + d) * Hi + h) * (size_t)Wi + w0 + w) * 32 + cq;
    uint2 o = {d0, d1};
    *(uint2*)(outp + obase) = o;
  }
}

// ---------------------------------------------------------------------------
// Implicit-GEMM 3x3x3 conv (pad 1, stride 1 or 2), bf16 interleaved in,
// f32 NCDHW out. One block = one (d_out, h_out, 32-wide W tile); wave mw
// handles cout tile mw, two 16x16 WMMA tiles (N = 32) per A fragment.
// Epilogue fuses BN scale/bias, residual add, ReLU.
// ---------------------------------------------------------------------------
#define CONV_WROWS 66   // >= 31*2+3 staged w rows
#define WSLAB_DW 6912   // max weight slab: 3*nCh(3)*MT(3)*256 dwords

__global__ void conv3d_wmma_kernel(const unsigned short* __restrict__ inBf,
                                   const uint32_t* __restrict__ wpk,
                                   float* __restrict__ outp,
                                   const float* __restrict__ scale,
                                   const float* __restrict__ bias,
                                   const float* __restrict__ resid,
                                   int Cin, int Cout,
                                   int Di, int Hi, int Wi,
                                   int Do, int Ho, int Wo,
                                   int stride, int relu) {
  __shared__ unsigned short sIn[CONV_WROWS * 32];   // [w][32c] bf16
#if HAVE_TDM
  __shared__ uint32_t sW[WSLAB_DW];
#endif
  int nCh = (Cin + 31) >> 5;
  int MT  = Cout >> 4;
  int lane = threadIdx.x & 31;
  int mw   = threadIdx.x >> 5;
  int nWT  = Wo >> 5;
  int b  = blockIdx.x;
  int wt = b % nWT;
  int hz = b / nWT;
  int ho = hz % Ho;
  int dz = hz / Ho;
  int w0 = wt << 5;

  int WST = 31 * stride + 3;           // staged w rows
  int wlo = w0 * stride - 1;           // staging origin

  v8f acc0 = {0.f, 0.f, 0.f, 0.f, 0.f, 0.f, 0.f, 0.f};
  v8f acc1 = {0.f, 0.f, 0.f, 0.f, 0.f, 0.f, 0.f, 0.f};

  int n  = lane & 15;                  // B fragment: N column
  int kb = (lane >> 4) << 3;           // B fragment: K base (0 or 8)

  for (int kd = 0; kd < 3; ++kd) {
    int zi = dz * stride + kd - 1;
    if (zi < 0 || zi >= Di) continue;
    for (int kh = 0; kh < 3; ++kh) {
      int yi = ho * stride + kh - 1;
      if (yi < 0 || yi >= Hi) continue;

#if HAVE_TDM
      // ---- TDM: DMA this (kd,kh) weight slab (3 kw * nCh * MT frags) to LDS
      if (threadIdx.x == 0) {
        int tapBase = (kd * 3 + kh) * 3;
        uint32_t Ndw = (uint32_t)(3 * nCh * MT * 256);
        uint64_t ga  = (uint64_t)(uintptr_t)(wpk + (size_t)tapBase * nCh * MT * 256);
        uint32_t la  = (uint32_t)(uintptr_t)&sW[0];
        u32x4 g0 = { 1u,                               // count=1, user desc
                     la,                               // lds_addr
                     (uint32_t)ga,                     // global_addr[31:0]
                     (uint32_t)((ga >> 32) & 0x01FFFFFFull) | (2u << 30) }; // type=2
        i32x8 g1 = { (int)(2u << 16),                  // data_size = 4B
                     (int)((Ndw & 0xFFFFu) << 16),     // tensor_dim0 lo
                     (int)(((Ndw >> 16) & 0xFFFFu) | (1u << 16)), // dim0 hi | dim1=1
                     (int)((Ndw & 0xFFFFu) << 16),     // tile_dim0
                     1,                                // tile_dim1 = 1
                     (int)Ndw,                         // tensor_dim0_stride lo
                     0, 0 };
        i32x4 g2 = {0, 0, 0, 0};
        i32x4 g3 = {0, 0, 0, 0};
#if __clang_major__ >= 23
        i32x8 g4 = {0, 0, 0, 0, 0, 0, 0, 0};
        __builtin_amdgcn_tensor_load_to_lds(g0, g1, g2, g3, g4, 0);
#else
        __builtin_amdgcn_tensor_load_to_lds(g0, g1, g2, g3, 0);
#endif
        __builtin_amdgcn_s_wait_tensorcnt(0);
      }
      __syncthreads();
#endif

      for (int ch = 0; ch < nCh; ++ch) {
        // ---- stage WST rows x 64B (32 bf16 channels) into LDS ----
        size_t rowBase = (((size_t)ch * Di + zi) * Hi + yi) * (size_t)Wi * 32;
        int chunksTot = WST * 4;                       // 16B chunks
        for (int p = threadIdx.x; p < chunksTot; p += blockDim.x) {
          int sidx = p >> 2;
          int qk   = (p & 3) << 3;                     // channel sub-offset
          int wi   = wlo + sidx;
          unsigned short* lp = &sIn[sidx * 32 + qk];
          if (wi >= 0 && wi < Wi) {
#if USE_ASYNC_STAGE
            uint32_t ldsOff = (uint32_t)(uintptr_t)lp;
            uint64_t gaddr  = (uint64_t)(uintptr_t)(inBf + rowBase + (size_t)wi * 32 + qk);
            asm volatile("global_load_async_to_lds_b128 %0, %1, off"
                         :: "v"(ldsOff), "v"(gaddr) : "memory");
#else
            *(uint4*)lp = *(const uint4*)(inBf + rowBase + (size_t)wi * 32 + qk);
#endif
          } else {
            uint4 z = {0, 0, 0, 0};
            *(uint4*)lp = z;
          }
        }
#if USE_ASYNC_STAGE
        asm volatile("s_wait_asynccnt 0x0" ::: "memory");
#endif
        __syncthreads();

        // ---- 3 taps (kw), 2 WMMAs each (N halves) ----
        for (int kw = 0; kw < 3; ++kw) {
#if HAVE_TDM
          const uint32_t* ap =
              &sW[(((size_t)kw * nCh + ch) * MT + mw) * 256 + (size_t)lane * 8];
#else
          int tap = (kd * 3 + kh) * 3 + kw;
          const uint32_t* ap =
              wpk + ((((size_t)tap * nCh + ch) * MT + mw) * 32 + lane) * 8;
#endif
          union Frag { uint4 q[2]; uint32_t u[8]; v16bf v; } A, B0, B1;
#pragma unroll
          for (int j = 0; j < 8; ++j) A.u[j] = ap[j];
          int wit0 = n * stride + kw;                  // staged row index
          int wit1 = wit0 + 16 * stride;
          const uint4* bp0 = (const uint4*)&sIn[wit0 * 32 + kb];
          const uint4* bp1 = (const uint4*)&sIn[wit1 * 32 + kb];
          B0.q[0] = bp0[0]; B0.q[1] = bp0[2];
          B1.q[0] = bp1[0]; B1.q[1] = bp1[2];
          acc0 = __builtin_amdgcn_wmma_f32_16x16x32_bf16(
              false, A.v, false, B0.v, (short)0, acc0, false, false);
          acc1 = __builtin_amdgcn_wmma_f32_16x16x32_bf16(
              false, A.v, false, B1.v, (short)0, acc1, false, false);
        }
        __syncthreads();
      }
    }
  }

  // ---- epilogue: C layout (32-bit 16x16): VGPR r -> M = r (+8 for hi lanes)
  int mbase = (lane >> 4) << 3;
#pragma unroll
  for (int r = 0; r < 8; ++r) {
    int co = mw * 16 + mbase + r;
    float sc = scale ? scale[co] : 1.f;
    float bi = scale ? bias[co] : 0.f;
    size_t idx0 = (((size_t)co * Do + dz) * Ho + ho) * (size_t)Wo + (w0 + n);
    size_t idx1 = idx0 + 16;
    float f0 = acc0[r] * sc + bi;
    float f1 = acc1[r] * sc + bi;
    if (resid) { f0 += resid[idx0]; f1 += resid[idx1]; }
    if (relu)  { f0 = fmaxf(f0, 0.f); f1 = fmaxf(f1, 0.f); }
    outp[idx0] = f0;
    outp[idx1] = f1;
  }
}

// ---------------------------------------------------------------------------
// Direct transposed conv (stride 2, k=3, "same*2" output), fused BN+ReLU.
// w_t layout (Cin, Cout, 3,3,3); conv weight = flip+transpose of w_t.
// ---------------------------------------------------------------------------
__global__ void deconv3d_kernel(const float* __restrict__ in,
                                const float* __restrict__ wt,
                                float* __restrict__ outp,
                                const float* __restrict__ scale,
                                const float* __restrict__ bias,
                                int Cin, int Cout,
                                int Di, int Hi, int Wi,
                                int Do, int Ho, int Wo, int relu) {
  int t = blockIdx.x * blockDim.x + threadIdx.x;
  int total = Cout * Do * Ho * Wo;
  if (t >= total) return;
  int x = t % Wo; int rest = t / Wo;
  int y = rest % Ho; rest /= Ho;
  int z = rest % Do; int co = rest / Do;

  int az[2], tz[2], nz = 0;
  int ay[2], ty[2], ny = 0;
  int ax[2], tx[2], nx = 0;
  for (int a = 0; a < 3; ++a) {
    int p = z + a - 1;
    if (!(p & 1) && p >= 0) { int q = p >> 1; if (q < Di) { az[nz] = a; tz[nz] = q; ++nz; } }
  }
  for (int a = 0; a < 3; ++a) {
    int p = y + a - 1;
    if (!(p & 1) && p >= 0) { int q = p >> 1; if (q < Hi) { ay[ny] = a; ty[ny] = q; ++ny; } }
  }
  for (int a = 0; a < 3; ++a) {
    int p = x + a - 1;
    if (!(p & 1) && p >= 0) { int q = p >> 1; if (q < Wi) { ax[nx] = a; tx[nx] = q; ++nx; } }
  }

  float acc = 0.f;
  for (int ci = 0; ci < Cin; ++ci) {
    const float* wb = wt + ((size_t)ci * Cout + co) * 27;
    const float* ib = in + (size_t)ci * Di * Hi * Wi;
    for (int i = 0; i < nz; ++i)
      for (int j = 0; j < ny; ++j)
        for (int k = 0; k < nx; ++k) {
          float wv = wb[(((2 - az[i]) * 3 + (2 - ay[j])) * 3) + (2 - ax[k])];
          acc += wv * ib[((size_t)tz[i] * Hi + ty[j]) * Wi + tx[k]];
        }
  }
  if (scale) acc = acc * scale[co] + bias[co];
  if (relu)  acc = fmaxf(acc, 0.f);
  outp[t] = acc;
}

// ---------------------------------------------------------------------------
// SGA directional scan. One wave handles 32/D columns; lane -> d.
// cur = k0*x + k1*prev + k2*prev[d-1] + k3*prev[d+1] + k4*max_d(prev)
// with l1-normalized 5-tap weights read per step. mode 0: write; 1: max.
// ---------------------------------------------------------------------------
__global__ void sga_scan_kernel(const float* __restrict__ x,
                                const float* __restrict__ g,
                                float* __restrict__ outp,
                                int C, int D, int H, int W,
                                int axis /*0=W,1=H*/, int rev, int mode) {
  int lane = threadIdx.x & 31;
  int waveInBlock = threadIdx.x >> 5;
  int cpw = 32 / D;
  int waveId = blockIdx.x * (blockDim.x >> 5) + waveInBlock;
  int colInWave = (cpw == 2) ? (lane >> 4) : 0;
  int d = lane & (D - 1);
  int col = waveId * cpw + colInWave;
  int qdim = (axis == 0) ? H : W;
  int ncols = C * qdim;
  bool active = (col < ncols);
  int c = active ? (col / qdim) : 0;
  int q = active ? (col % qdim) : 0;
  int L = (axis == 0) ? W : H;
  int p0 = rev ? (L - 1) : 0;
  int dp = rev ? -1 : 1;

  size_t HW = (size_t)H * W;
  size_t xbase = ((size_t)c * D + d) * HW;
  size_t kbase = (size_t)c * 5 * HW;

  float st = 0.f;
  for (int i = 0; i < L; ++i) {
    int p = p0 + dp * i;
    int hh = (axis == 0) ? q : p;
    int ww = (axis == 0) ? p : q;
    size_t xidx = xbase + (size_t)hh * W + ww;
    float xi = active ? x[xidx] : 0.f;
    float cur;
    if (i == 0) {
      cur = xi;
    } else {
      float k0 = 0.f, k1 = 0.f, k2 = 0.f, k3 = 0.f, k4 = 0.f;
      if (active) {
        size_t kk = kbase + (size_t)hh * W + ww;
        k0 = g[kk];          k1 = g[kk + HW];
        k2 = g[kk + 2 * HW]; k3 = g[kk + 3 * HW];
        k4 = g[kk + 4 * HW];
      }
      float s = fabsf(k0) + fabsf(k1) + fabsf(k2) + fabsf(k3) + fabsf(k4);
      float inv = 1.f / fmaxf(s, 1e-12f);
      k0 *= inv; k1 *= inv; k2 *= inv; k3 *= inv; k4 *= inv;
      float pdm = __shfl_up(st, 1, D);   if (d == 0)     pdm = 0.f;
      float pdp = __shfl_down(st, 1, D); if (d == D - 1) pdp = 0.f;
      float pm = st;
      for (int off = D >> 1; off > 0; off >>= 1)
        pm = fmaxf(pm, __shfl_xor(pm, off, D));
      cur = k0 * xi + k1 * st + k2 * pdm + k3 * pdp + k4 * pm;
    }
    st = cur;
    if (active) {
      if (mode == 0) outp[xidx] = cur;
      else           outp[xidx] = fmaxf(outp[xidx], cur);
    }
  }
}

// ---------------------------------------------------------------------------
// Elementwise helpers
// ---------------------------------------------------------------------------
__global__ void affine_kernel(const float* __restrict__ in, float* __restrict__ outp,
                              const float* __restrict__ scale,
                              const float* __restrict__ bias,
                              int plane, int relu, int total) {
  int t = blockIdx.x * blockDim.x + threadIdx.x;
  if (t >= total) return;
  int c = t / plane;
  float f = in[t] * scale[c] + bias[c];
  if (relu) f = fmaxf(f, 0.f);
  outp[t] = f;
}

__global__ void make_sb_kernel(const float* g, const float* b,
                               const float* m, const float* v,
                               int C, float* sb) {
  int t = threadIdx.x;
  if (t < C) {
    float s = g[t] * rsqrtf(v[t] + 1e-5f);
    sb[t] = s;
    sb[128 + t] = b[t] - m[t] * s;
  }
}

// ===========================================================================
// Host orchestration
// ===========================================================================
// Input layout assumption (jax pytree flattening; dicts in sorted-key order):
enum {
  IN_X = 0, IN_SG1, IN_SG11, IN_SG12, IN_SG2,
  P_CONV1A_BN = 5,  P_CONV1A_W = 9,
  P_CONV2A_BN = 10, P_CONV2A_W = 14,
  P_CONV_END_W = 15,
  P_CONV_START_BN = 16, P_CONV_START_W = 20,
  P_DC1A_BN1 = 21, P_DC1A_BN2 = 25, P_DC1A_W1 = 29, P_DC1A_W2 = 30,
  P_DC2A_BN1 = 31, P_DC2A_BN2 = 35, P_DC2A_W1 = 39, P_DC2A_W2 = 40,
  P_SGA1_BN1 = 41,  P_SGA1_BN2 = 45,  P_SGA1_W = 49,
  P_SGA11_BN1 = 50, P_SGA11_BN2 = 54, P_SGA11_W = 58,
  P_SGA12_BN1 = 59, P_SGA12_BN2 = 63, P_SGA12_W = 67,
  P_SGA2_BN1 = 68,  P_SGA2_BN2 = 72,  P_SGA2_W = 76,
};

static inline int ceilDiv(int a, int b) { return (a + b - 1) / b; }

extern "C" void kernel_launch(void* const* d_in, const int* in_sizes, int n_in,
                              void* d_out, int out_size, void* d_ws, size_t ws_size,
                              hipStream_t stream) {
  (void)in_sizes; (void)n_in; (void)out_size;
  auto F = [&](int i) { return (const float*)d_in[i]; };

  // dims
  const int FD = 32, FH = 64, FW = 128;        // full res
  const int HD = 16, HH = 32, HWd = 64;        // half res
  const int QD = 8,  QH = 16, QW = 32;         // quarter res
  const int P1 = FD * FH * FW;                 // 262144
  const int P2 = HD * HH * HWd;                // 32768
  const int P3 = QD * QH * QW;                 // 4096

  // ---- workspace layout (bump allocator) ----
  uint8_t* base = (uint8_t*)d_ws;
  size_t off = 0;
  auto alloc = [&](size_t bytes) -> void* {
    void* p = base + off;
    off += (bytes + 255) & ~(size_t)255;
    return p;
  };
  float* A1  = (float*)alloc((size_t)32 * P1 * 4);
  float* A2  = (float*)alloc((size_t)32 * P1 * 4);
  float* C64 = (float*)alloc((size_t)64 * P1 * 4);   // scan tmp / concat (full)
  float* Hb1 = (float*)alloc((size_t)48 * P2 * 4);
  float* Hb2 = (float*)alloc((size_t)48 * P2 * 4);
  float* Hb3 = (float*)alloc((size_t)48 * P2 * 4);
  float* C96 = (float*)alloc((size_t)96 * P2 * 4);
  float* Qb  = (float*)alloc((size_t)64 * P3 * 4);
  unsigned short* BF = (unsigned short*)alloc((size_t)64 * P1 * 2);

  auto wpkDW = [](int Cin, int Cout) {
    return (size_t)27 * ((Cin + 31) / 32) * (Cout / 16) * 32 * 8;
  };
  uint32_t* wpkCS   = (uint32_t*)alloc(wpkDW(64, 32) * 4);
  uint32_t* wpkS1   = (uint32_t*)alloc(wpkDW(32, 32) * 4);
  uint32_t* wpkC1A  = (uint32_t*)alloc(wpkDW(32, 48) * 4);
  uint32_t* wpkS11  = (uint32_t*)alloc(wpkDW(48, 48) * 4);
  uint32_t* wpkC2A  = (uint32_t*)alloc(wpkDW(48, 64) * 4);
  uint32_t* wpkD2W2 = (uint32_t*)alloc(wpkDW(96, 48) * 4);
  uint32_t* wpkS12  = (uint32_t*)alloc(wpkDW(48, 48) * 4);
  uint32_t* wpkD1W2 = (uint32_t*)alloc(wpkDW(64, 32) * 4);
  uint32_t* wpkS2   = (uint32_t*)alloc(wpkDW(32, 32) * 4);
  uint32_t* wpkCE   = (uint32_t*)alloc(wpkDW(32, 64) * 4);

  auto sbAlloc = [&]() { return (float*)alloc(256 * 4); };
  float* sbCS   = sbAlloc(); float* sbC1A  = sbAlloc(); float* sbC2A = sbAlloc();
  float* sbS1a  = sbAlloc(); float* sbS1b  = sbAlloc();
  float* sbS11a = sbAlloc(); float* sbS11b = sbAlloc();
  float* sbS12a = sbAlloc(); float* sbS12b = sbAlloc();
  float* sbS2a  = sbAlloc(); float* sbS2b  = sbAlloc();
  float* sbD2a  = sbAlloc(); float* sbD2b  = sbAlloc();
  float* sbD1a  = sbAlloc(); float* sbD1b  = sbAlloc();
  if (ws_size < off) return;  // not enough scratch; nothing sane to do

  // ---- scale/bias from BN params (leaves sorted: b,g,m,v) ----
  auto mkSB = [&](int idx, int C, float* dst) {
    make_sb_kernel<<<1, 128, 0, stream>>>(F(idx + 1), F(idx + 0), F(idx + 2),
                                          F(idx + 3), C, dst);
  };
  mkSB(P_CONV_START_BN, 32, sbCS);
  mkSB(P_CONV1A_BN, 48, sbC1A);
  mkSB(P_CONV2A_BN, 64, sbC2A);
  mkSB(P_SGA1_BN1, 32, sbS1a);   mkSB(P_SGA1_BN2, 32, sbS1b);
  mkSB(P_SGA11_BN1, 48, sbS11a); mkSB(P_SGA11_BN2, 48, sbS11b);
  mkSB(P_SGA12_BN1, 48, sbS12a); mkSB(P_SGA12_BN2, 48, sbS12b);
  mkSB(P_SGA2_BN1, 32, sbS2a);   mkSB(P_SGA2_BN2, 32, sbS2b);
  mkSB(P_DC2A_BN1, 48, sbD2a);   mkSB(P_DC2A_BN2, 48, sbD2b);
  mkSB(P_DC1A_BN1, 32, sbD1a);   mkSB(P_DC1A_BN2, 32, sbD1b);

  // ---- pre-pack all WMMA conv weights ----
  auto pp = [&](int widx, int Cin, int Cout, uint32_t* dst) {
    int total = 27 * ((Cin + 31) / 32) * (Cout / 16) * 32;
    prepack_w_kernel<<<ceilDiv(total, 256), 256, 0, stream>>>(F(widx), dst, Cin, Cout);
  };
  pp(P_CONV_START_W, 64, 32, wpkCS);
  pp(P_SGA1_W,  32, 32, wpkS1);
  pp(P_CONV1A_W, 32, 48, wpkC1A);
  pp(P_SGA11_W, 48, 48, wpkS11);
  pp(P_CONV2A_W, 48, 64, wpkC2A);
  pp(P_DC2A_W2, 96, 48, wpkD2W2);
  pp(P_SGA12_W, 48, 48, wpkS12);
  pp(P_DC1A_W2, 64, 32, wpkD1W2);
  pp(P_SGA2_W,  32, 32, wpkS2);
  pp(P_CONV_END_W, 32, 64, wpkCE);

  // ---- helpers ----
  auto cvt = [&](const float* src, int C, int Dd, int Hh, int Ww) {
    int nCb = (C + 31) / 32;
    int grid = nCb * Dd * Hh * (Ww / 32);
    cvt_interleave_kernel<<<grid, 256, 0, stream>>>(src, BF, C, Dd, Hh, Ww);
  };
  auto conv = [&](const uint32_t* wpk, float* out, int Cin, int Cout,
                  int Di, int Hi, int Wi, int stride,
                  const float* sb, const float* res, int relu) {
    int Do = Di / stride, Ho = Hi / stride, Wo = Wi / stride;
    dim3 grid(Do * Ho * (Wo / 32)), blk(32 * (Cout / 16));
    conv3d_wmma_kernel<<<grid, blk, 0, stream>>>(
        BF, wpk, out, sb, sb ? sb + 128 : nullptr, res,
        Cin, Cout, Di, Hi, Wi, Do, Ho, Wo, stride, relu);
  };
  auto sga = [&](const float* cur, const float* g, float* tmp,
                 int C, int D, int Hh, int Ww) {
    size_t part = (size_t)5 * C * Hh * Ww;
    auto scan = [&](int axis, int rev, int mode, const float* gp) {
      int nc = C * ((axis == 0) ? Hh : Ww);
      int cpw = 32 / D;
      int waves = ceilDiv(nc, cpw);
      sga_scan_kernel<<<ceilDiv(waves, 4), 128, 0, stream>>>(
          cur, gp, tmp, C, D, Hh, Ww, axis, rev, mode);
    };
    scan(0, 0, 0, g + 0 * part);
    scan(0, 1, 1, g + 1 * part);
    scan(1, 0, 1, g + 2 * part);
    scan(1, 1, 1, g + 3 * part);
  };
  auto affine = [&](float* buf, const float* sb, int plane, int C) {
    int total = C * plane;
    affine_kernel<<<ceilDiv(total, 256), 256, 0, stream>>>(
        buf, buf, sb, sb + 128, plane, 1, total);
  };

  // ================= pipeline =================
  // conv_start + BN (no relu)
  cvt(F(IN_X), 64, FD, FH, FW);
  conv(wpkCS, A1, 64, 32, FD, FH, FW, 1, sbCS, nullptr, 0);

  // sga_block 1 (C=32, full res)
  sga(A1, F(IN_SG1), C64, 32, 32, FH, FW);
  affine(C64, sbS1a, P1, 32);
  cvt(C64, 32, FD, FH, FW);
  conv(wpkS1, A2, 32, 32, FD, FH, FW, 1, sbS1b, A1, 1);   // rem0 = A2

  // conv1a (stride 2) + BN + relu
  cvt(A2, 32, FD, FH, FW);
  conv(wpkC1A, Hb1, 32, 48, FD, FH, FW, 2, sbC1A, nullptr, 1);

  // sga_block 11 (C=48, half res, D=16)
  sga(Hb1, F(IN_SG11), Hb3, 48, 16, HH, HWd);
  affine(Hb3, sbS11a, P2, 48);
  cvt(Hb3, 48, HD, HH, HWd);
  conv(wpkS11, Hb2, 48, 48, HD, HH, HWd, 1, sbS11b, Hb1, 1);  // rem1 = Hb2

  // conv2a (stride 2) + BN + relu
  cvt(Hb2, 48, HD, HH, HWd);
  conv(wpkC2A, Qb, 48, 64, HD, HH, HWd, 2, sbC2A, nullptr, 1);

  // deconv2a_w1 (64->48, x2 up) + BN + relu -> concat channels 0..47
  deconv3d_kernel<<<ceilDiv(48 * P2, 256), 256, 0, stream>>>(
      Qb, F(P_DC2A_W1), C96, sbD2a, sbD2a + 128,
      64, 48, QD, QH, QW, HD, HH, HWd, 1);
  hipMemcpyAsync(C96 + (size_t)48 * P2, Hb2, (size_t)48 * P2 * 4,
                 hipMemcpyDeviceToDevice, stream);
  cvt(C96, 96, HD, HH, HWd);
  conv(wpkD2W2, Hb1, 96, 48, HD, HH, HWd, 1, sbD2b, nullptr, 1);

  // sga_block 12
  sga(Hb1, F(IN_SG12), Hb3, 48, 16, HH, HWd);
  affine(Hb3, sbS12a, P2, 48);
  cvt(Hb3, 48, HD, HH, HWd);
  conv(wpkS12, Hb2, 48, 48, HD, HH, HWd, 1, sbS12b, Hb1, 1);

  // deconv1a_w1 (48->32, x2 up) + BN + relu -> concat channels 0..31
  deconv3d_kernel<<<ceilDiv(32 * P1, 256), 256, 0, stream>>>(
      Hb2, F(P_DC1A_W1), C64, sbD1a, sbD1a + 128,
      48, 32, HD, HH, HWd, FD, FH, FW, 1);
  hipMemcpyAsync(C64 + (size_t)32 * P1, A2, (size_t)32 * P1 * 4,
                 hipMemcpyDeviceToDevice, stream);
  cvt(C64, 64, FD, FH, FW);
  conv(wpkD1W2, A1, 64, 32, FD, FH, FW, 1, sbD1b, nullptr, 0);  // BN, no relu

  // sga_block 2
  sga(A1, F(IN_SG2), C64, 32, 32, FH, FW);
  affine(C64, sbS2a, P1, 32);
  cvt(C64, 32, FD, FH, FW);
  conv(wpkS2, A2, 32, 32, FD, FH, FW, 1, sbS2b, A1, 1);

  // conv_end (32->64) + relu, no BN -> d_out
  cvt(A2, 32, FD, FH, FW);
  conv(wpkCE, (float*)d_out, 32, 64, FD, FH, FW, 1, nullptr, nullptr, 1);
}
